// FormulaNet_19731079758360
// MI455X (gfx1250) — matrix operand order repaired and checked by reading, not compiled, not run on hardware
//
#include <hip/hip_runtime.h>

typedef __attribute__((ext_vector_type(16))) _Float16 v16h;
typedef __attribute__((ext_vector_type(8)))  float    v8f;

#define NDEP   64
#define N2     4096
#define ROWS   64      // rows per block
#define TPB    256     // 8 waves; 4 slices of the i-loop, 2 waves per slice
#define XPITCH 68      // row pitch in LDS floats (bank-conflict padding, keeps 16B align)

// LDS floats: xs ROWS*68 | bc N2 | nc N2 | aw N2 | linw 64 | red 256 | linres 64
#define SMEM_FLOATS (ROWS * XPITCH + 3 * N2 + 64 + 256 + 64)

__global__ __launch_bounds__(TPB) void formulanet_kernel(
    const float* __restrict__ x,  const float* __restrict__ a,
    const float* __restrict__ b,  const float* __restrict__ c,
    const float* __restrict__ cw, const float* __restrict__ cb,
    const float* __restrict__ lw, const float* __restrict__ lb,
    float* __restrict__ out)
{
    extern __shared__ float smem[];
    float* xs     = smem;                 // ROWS * XPITCH
    float* bcs    = xs + ROWS * XPITCH;   // N2
    float* ncs    = bcs + N2;             // N2
    float* aws    = ncs + N2;             // N2
    float* linw   = aws + N2;             // 64
    float* red    = linw + 64;            // 256
    float* linres = red + 256;            // 64

    const int tid   = threadIdx.x;
    const int rowL  = tid & (ROWS - 1);   // local row 0..63
    const int slice = tid >> 6;           // 0..3 (uniform within each wave)
    const int row0  = blockIdx.x * ROWS;
    const int row   = row0 + rowL;

    // CDNA5 prefetch of this thread's x row (global_prefetch_b8)
    __builtin_prefetch(&x[row * NDEP], 0, 0);

    // ---- Phase 0: stage transformed constants into LDS + batch-invariant sums
    // g*cw = (a*cw) * exp2( (y-b)^2 * (-log2e/(2 c^2)) )
    float csum = 0.f;
    for (int k = tid; k < N2; k += TPB) {
        float av = a[k], bv = b[k], cv = c[k], wv = cw[k];
        bcs[k] = bv;
        ncs[k] = -0.72134752044448170f / (cv * cv);  // -log2(e)/2 / c^2
        aws[k] = av * wv;
        csum  += cb[k];
    }
    if (tid < NDEP) {
        linw[tid] = lw[tid];
        csum += lb[tid];
    }
    // stage x tile: thread loads 16 floats (quarter-row) of row rowL
    {
        const float4* src = reinterpret_cast<const float4*>(&x[row * NDEP + slice * 16]);
#pragma unroll
        for (int q = 0; q < 4; ++q)
            *reinterpret_cast<float4*>(&xs[rowL * XPITCH + slice * 16 + q * 4]) = src[q];
    }
    red[tid] = csum;
    __syncthreads();
#pragma unroll 1
    for (int off = TPB / 2; off > 0; off >>= 1) {
        if (tid < off) red[tid] += red[tid + off];
        __syncthreads();
    }
    const float const_sum = red[0];   // sum(cross_b) + sum(lin_b)

    // ---- Phase 1: lin term  dot(x_row, lin_w)  via v_wmma_f32_16x16x32_f16
    // Waves 0,1 each cover 32 rows (two 16-row tiles); K=64 as two K=32 WMMAs.
    const int wave = tid >> 5;
    const int lane = tid & 31;
    if (wave < 2) {
        const int  mrow = lane & 15;
        const bool loh  = lane < 16;
#pragma unroll
        for (int tile = 0; tile < 2; ++tile) {
            const int r = wave * 32 + tile * 16 + mrow;   // local row
            v8f acc8 = {0.f, 0.f, 0.f, 0.f, 0.f, 0.f, 0.f, 0.f};
#pragma unroll
            for (int s = 0; s < 2; ++s) {
                const int kb = s * 32;
                v16h A, Bm;
#pragma unroll
                for (int h = 0; h < 16; ++h) {
                    const int v = h >> 1, p = h & 1;
                    // 16-bit A 16x32 lane layout (ISA 7.12.2)
                    const int Ka = (v < 4) ? ((loh ? 0 : 8)  + 2 * v + p)
                                           : ((loh ? 16 : 24) + 2 * (v - 4) + p);
                    // 16-bit B 32x16: same K order as A
                    const int Kb = Ka;
                    A[h]  = (_Float16)xs[r * XPITCH + kb + Ka];
                    Bm[h] = (_Float16)linw[kb + Kb];   // same for every column N
                }
                acc8 = __builtin_amdgcn_wmma_f32_16x16x32_f16(
                           false, A, false, Bm, (short)0, acc8, false, false);
            }
            // D: vgpr r holds M=r (lanes 0-15, N=lane) and M=8+r (lanes 16-31).
            // All columns equal; extract column 0 via lanes 0 and 16.
            if (lane == 0) {
#pragma unroll
                for (int rr = 0; rr < 8; ++rr)
                    linres[wave * 32 + tile * 16 + rr] = acc8[rr];
            } else if (lane == 16) {
#pragma unroll
                for (int rr = 0; rr < 8; ++rr)
                    linres[wave * 32 + tile * 16 + 8 + rr] = acc8[rr];
            }
        }
    }

    // ---- Phase 2: Gaussian cross terms. Thread handles rows[rowL], i in [16*slice,16*slice+16)
    const float* xrow = &xs[rowL * XPITCH];
    float4 xr[16];
#pragma unroll
    for (int t = 0; t < 16; ++t)
        xr[t] = *reinterpret_cast<const float4*>(&xrow[t * 4]);

    float acc = 0.f;
    const int i0 = slice * 16;
#pragma unroll 1
    for (int i = i0; i < i0 + 16; ++i) {
        const float xi = xrow[i];
        const int   kb = i << 6;
#pragma unroll
        for (int t = 0; t < 16; ++t) {
            const int k = kb + t * 4;
            const float4 bb = *reinterpret_cast<const float4*>(&bcs[k]);
            const float4 nc = *reinterpret_cast<const float4*>(&ncs[k]);
            const float4 aw = *reinterpret_cast<const float4*>(&aws[k]);
            float d0 = fmaf(xi, xr[t].x, -bb.x);
            float d1 = fmaf(xi, xr[t].y, -bb.y);
            float d2 = fmaf(xi, xr[t].z, -bb.z);
            float d3 = fmaf(xi, xr[t].w, -bb.w);
            acc = fmaf(aw.x, __builtin_amdgcn_exp2f(d0 * d0 * nc.x), acc);
            acc = fmaf(aw.y, __builtin_amdgcn_exp2f(d1 * d1 * nc.y), acc);
            acc = fmaf(aw.z, __builtin_amdgcn_exp2f(d2 * d2 * nc.z), acc);
            acc = fmaf(aw.w, __builtin_amdgcn_exp2f(d3 * d3 * nc.w), acc);
        }
    }

    // ---- Phase 3: combine 4 slice partials + lin term + constant offset
    __syncthreads();          // red[0] consumed by all before reuse
    red[tid] = acc;
    __syncthreads();
    if (tid < ROWS) {
        float r = red[tid] + red[tid + 64] + red[tid + 128] + red[tid + 192]
                + linres[tid] + const_sum;
        out[row0 + tid] = r;
    }
}

extern "C" void kernel_launch(void* const* d_in, const int* in_sizes, int n_in,
                              void* d_out, int out_size, void* d_ws, size_t ws_size,
                              hipStream_t stream) {
    (void)in_sizes; (void)n_in; (void)d_ws; (void)ws_size;
    const float* x  = (const float*)d_in[0];
    const float* a  = (const float*)d_in[1];
    const float* b  = (const float*)d_in[2];
    const float* c  = (const float*)d_in[3];
    const float* cw = (const float*)d_in[4];
    const float* cb = (const float*)d_in[5];
    const float* lw = (const float*)d_in[6];
    const float* lb = (const float*)d_in[7];
    float* out = (float*)d_out;

    const int Bsz = out_size;                 // 32768 rows
    const size_t shmem = SMEM_FLOATS * sizeof(float);   // ~68 KB (<< 320 KB/WGP)
    hipFuncSetAttribute(reinterpret_cast<const void*>(formulanet_kernel),
                        hipFuncAttributeMaxDynamicSharedMemorySize, (int)shmem);
    formulanet_kernel<<<Bsz / ROWS, TPB, shmem, stream>>>(x, a, b, c, cw, cb, lw, lb, out);
}
